// SwiGLUExperts_27848567947311
// MI455X (gfx1250) — matrix operand order repaired and checked by reading, not compiled, not run on hardware
//
#include <hip/hip_runtime.h>
#include <hip/hip_bf16.h>

// ---------------------------------------------------------------------------
// MoE SwiGLU for MI455X (gfx1250, wave32):
//   kernel 1: hidden[t, 0:512]  = silu(x@Wg) * (x@Wu)   (bf16 out to d_ws)
//   kernel 2: out[t, 0:1024]    = hidden @ Wd           (f32 out to d_out)
// v_wmma_f32_16x16x32_bf16 with f32 accumulation. Double-buffered LDS
// software pipeline: next chunk's global loads (regs, or async->LDS for the
// bf16 A tiles in kernel 2) are in flight under the current chunk's WMMAs;
// one barrier per K-chunk.
// ---------------------------------------------------------------------------

#define USE_ASYNC_LDS 1

typedef __bf16 bf16;
typedef bf16  v16bf __attribute__((ext_vector_type(16)));
typedef bf16  v8bf  __attribute__((ext_vector_type(8)));
typedef float v8f   __attribute__((ext_vector_type(8)));
typedef __attribute__((address_space(3))) bf16* lds_bf16_t;

#define T_TOK 4096
#define H_DIM 1024
#define I_DIM 512
#define N_EXP 16

#define BM 64          // rows per block tile
#define BN 64          // cols per block tile
#define BK 32          // K per chunk (one wmma K)
#define BKP 40         // padded K stride in elements (80 bytes, 16B aligned)

__device__ __forceinline__ bf16 f2bf(float f) { return (bf16)f; }

__device__ __forceinline__ v16bf frag16(const bf16* p0, const bf16* p1) {
  v8bf a = *(const v8bf*)p0;     // ds_load_b128
  v8bf b = *(const v8bf*)p1;     // ds_load_b128
  return __builtin_shufflevector(a, b, 0, 1, 2, 3, 4, 5, 6, 7,
                                       8, 9, 10, 11, 12, 13, 14, 15);
}

// Map a flat row-tile index onto (expert, rowStart, rowsValid); tiles are
// aligned to each expert's token range so one tile never spans two experts.
__device__ __forceinline__ bool map_tile(const int* __restrict__ counts, int bm,
                                         int& rowStart, int& rowsValid, int& expert) {
  int acc = 0, tok = 0; expert = -1;
  rowStart = 0; rowsValid = 0;
#pragma unroll
  for (int i = 0; i < N_EXP; ++i) {
    int c  = counts[i];
    int nt = (c + BM - 1) / BM;
    if (expert < 0 && bm < acc + nt) {
      expert = i;
      int lt = bm - acc;
      rowStart  = tok + lt * BM;
      rowsValid = c - lt * BM;
      if (rowsValid > BM) rowsValid = BM;
    }
    acc += nt; tok += c;
  }
  return expert >= 0;
}

// ------------------------------- kernel 1 ----------------------------------
__global__ __launch_bounds__(256) void k_gate_up_silu(
    const float* __restrict__ x,          // [T, H]
    const float* __restrict__ wgu,        // [E, H, 2I]
    const int*   __restrict__ counts,     // [E]
    bf16*        __restrict__ hidden) {   // [T, I] (d_ws)
  __shared__ alignas(16) bf16 As[2][BM][BKP];
  __shared__ alignas(16) bf16 Bg[2][BN][BKP];  // transposed [N][K]
  __shared__ alignas(16) bf16 Bu[2][BN][BKP];

  int rowStart, rowsValid, e;
  if (!map_tile(counts, (int)blockIdx.y, rowStart, rowsValid, e)) return;

  const int tid   = threadIdx.x;
  const int lane  = tid & 31;
  const int wave  = tid >> 5;          // 8 waves
  const int mi2   = wave >> 2;         // 0..1 -> two 16-row subtiles each
  const int ni    = wave & 3;          // 0..3 (col subtile)
  const int col16 = lane & 15;
  const int hi    = (lane >> 4) & 1;
  const int colBase = blockIdx.x * BN; // within I

  const float* W = wgu + (size_t)e * H_DIM * (2 * I_DIM);

  // ---- per-thread staging geometry (fixed across chunks) ----
  const int rA = tid >> 3, sA = tid & 7;     // A: two rows rA, rA+32; float4 seg
  int tA0 = rowStart + rA;      if (tA0 > T_TOK - 1) tA0 = T_TOK - 1;
  int tA1 = rowStart + rA + 32; if (tA1 > T_TOK - 1) tA1 = T_TOK - 1;
  const float* xp0 = x + (size_t)tA0 * H_DIM + sA * 4;
  const float* xp1 = x + (size_t)tA1 * H_DIM + sA * 4;
  const int kB = tid >> 4;                   // B: rows kB, kB+16; cols nB..nB+3
  const int nB = (tid & 15) * 4;
  const float* Wb = W + colBase + nB;

  float4 a0, a1, g0, u0, g1, u1;             // in-flight registers

  auto load_regs = [&](int k0) {
    a0 = *(const float4*)(xp0 + k0);
    a1 = *(const float4*)(xp1 + k0);
    const float* p0 = Wb + (size_t)(k0 + kB) * (2 * I_DIM);
    const float* p1 = Wb + (size_t)(k0 + kB + 16) * (2 * I_DIM);
    g0 = *(const float4*)p0;  u0 = *(const float4*)(p0 + I_DIM);
    g1 = *(const float4*)p1;  u1 = *(const float4*)(p1 + I_DIM);
  };
  auto store_lds = [&](int buf) {
    bf16* d0 = &As[buf][rA][sA * 4];
    d0[0] = f2bf(a0.x); d0[1] = f2bf(a0.y); d0[2] = f2bf(a0.z); d0[3] = f2bf(a0.w);
    bf16* d1 = &As[buf][rA + 32][sA * 4];
    d1[0] = f2bf(a1.x); d1[1] = f2bf(a1.y); d1[2] = f2bf(a1.z); d1[3] = f2bf(a1.w);
    Bg[buf][nB + 0][kB] = f2bf(g0.x); Bg[buf][nB + 1][kB] = f2bf(g0.y);
    Bg[buf][nB + 2][kB] = f2bf(g0.z); Bg[buf][nB + 3][kB] = f2bf(g0.w);
    Bu[buf][nB + 0][kB] = f2bf(u0.x); Bu[buf][nB + 1][kB] = f2bf(u0.y);
    Bu[buf][nB + 2][kB] = f2bf(u0.z); Bu[buf][nB + 3][kB] = f2bf(u0.w);
    Bg[buf][nB + 0][kB + 16] = f2bf(g1.x); Bg[buf][nB + 1][kB + 16] = f2bf(g1.y);
    Bg[buf][nB + 2][kB + 16] = f2bf(g1.z); Bg[buf][nB + 3][kB + 16] = f2bf(g1.w);
    Bu[buf][nB + 0][kB + 16] = f2bf(u1.x); Bu[buf][nB + 1][kB + 16] = f2bf(u1.y);
    Bu[buf][nB + 2][kB + 16] = f2bf(u1.z); Bu[buf][nB + 3][kB + 16] = f2bf(u1.w);
  };

  v8f accG[2] = {{}, {}};
  v8f accU[2] = {{}, {}};

  auto compute = [&](int buf) {
    const bf16* pbg = &Bg[buf][ni * 16 + col16][hi ? 16 : 0];
    v16bf bg = frag16(pbg, pbg + 8);
    const bf16* pbu = &Bu[buf][ni * 16 + col16][hi ? 16 : 0];
    v16bf bu = frag16(pbu, pbu + 8);
#pragma unroll
    for (int sub = 0; sub < 2; ++sub) {
      const bf16* pa = &As[buf][mi2 * 32 + sub * 16 + col16][hi ? 16 : 0];
      v16bf a = frag16(pa, pa + 8);
      accG[sub] = __builtin_amdgcn_wmma_f32_16x16x32_bf16(
          false, a, false, bg, (short)0, accG[sub], false, false);
      accU[sub] = __builtin_amdgcn_wmma_f32_16x16x32_bf16(
          false, a, false, bu, (short)0, accU[sub], false, false);
    }
  };

  // ---- pipelined main loop: one barrier per chunk ----
  load_regs(0);
  store_lds(0);
  __syncthreads();
  const int nk = H_DIM / BK;
  for (int k = 0; k < nk; ++k) {
    const int cur = k & 1;
    if (k + 1 < nk) {
      load_regs((k + 1) * BK);           // global loads in flight under wmma
      __builtin_prefetch(Wb + (size_t)((k + 2) * BK + kB) * (2 * I_DIM), 0, 0);
    }
    compute(cur);
    if (k + 1 < nk) store_lds(cur ^ 1);
    __syncthreads();
  }

  // ---- h = silu(G) * U, store bf16 ----
#pragma unroll
  for (int sub = 0; sub < 2; ++sub) {
#pragma unroll
    for (int r = 0; r < 8; ++r) {
      int m = mi2 * 32 + sub * 16 + r + hi * 8;   // C/D row layout
      if (m < rowsValid) {
        float g = accG[sub][r];
        float u = accU[sub][r];
        float h = (g / (1.0f + __expf(-g))) * u;
        hidden[(size_t)(rowStart + m) * I_DIM + colBase + ni * 16 + col16] = f2bf(h);
      }
    }
  }
}

// ------------------------------- kernel 2 ----------------------------------
__global__ __launch_bounds__(256) void k_down(
    const bf16*  __restrict__ hidden,    // [T, I] bf16 (d_ws)
    const float* __restrict__ wd,        // [E, I, H]
    const int*   __restrict__ counts,    // [E]
    float*       __restrict__ out) {     // [T, H]
  __shared__ alignas(16) bf16 As[2][BM][BKP];
  __shared__ alignas(16) bf16 Bs[2][BN][BKP];  // transposed [N][K]

  int rowStart, rowsValid, e;
  if (!map_tile(counts, (int)blockIdx.y, rowStart, rowsValid, e)) return;

  const int tid   = threadIdx.x;
  const int lane  = tid & 31;
  const int wave  = tid >> 5;
  const int mi2   = wave >> 2;
  const int ni    = wave & 3;
  const int col16 = lane & 15;
  const int hi    = (lane >> 4) & 1;
  const int colBase = blockIdx.x * BN;   // within H

  const float* W = wd + (size_t)e * I_DIM * H_DIM;

  // ---- staging geometry ----
  const int rA = tid >> 2, sA = tid & 3;       // A: 16B per thread per chunk
  int tA = rowStart + rA; if (tA > T_TOK - 1) tA = T_TOK - 1;
  const bf16* hp = hidden + (size_t)tA * I_DIM + sA * 8;
  const int kB = tid >> 4;                     // B: rows kB, kB+16
  const int nB = (tid & 15) * 4;
  const float* Wb = W + colBase + nB;

  auto issue_A = [&](int k0, int buf) {        // async global -> LDS (bf16 copy)
#if USE_ASYNC_LDS
    unsigned ldsOff = (unsigned)(unsigned long long)(lds_bf16_t)&As[buf][rA][sA * 8];
    asm volatile("global_load_async_to_lds_b128 %0, %1, off"
                 :: "v"(ldsOff), "v"((unsigned long long)(hp + k0)) : "memory");
#else
    *(v8bf*)&As[buf][rA][sA * 8] = *(const v8bf*)(hp + k0);
#endif
  };

  float4 b0, b1;
  auto load_B = [&](int k0) {
    b0 = *(const float4*)(Wb + (size_t)(k0 + kB) * H_DIM);
    b1 = *(const float4*)(Wb + (size_t)(k0 + kB + 16) * H_DIM);
  };
  auto store_B = [&](int buf) {
    Bs[buf][nB + 0][kB] = f2bf(b0.x); Bs[buf][nB + 1][kB] = f2bf(b0.y);
    Bs[buf][nB + 2][kB] = f2bf(b0.z); Bs[buf][nB + 3][kB] = f2bf(b0.w);
    Bs[buf][nB + 0][kB + 16] = f2bf(b1.x); Bs[buf][nB + 1][kB + 16] = f2bf(b1.y);
    Bs[buf][nB + 2][kB + 16] = f2bf(b1.z); Bs[buf][nB + 3][kB + 16] = f2bf(b1.w);
  };

  v8f acc[2] = {{}, {}};

  auto compute = [&](int buf) {
    const bf16* pb = &Bs[buf][ni * 16 + col16][hi ? 16 : 0];
    v16bf b = frag16(pb, pb + 8);
#pragma unroll
    for (int sub = 0; sub < 2; ++sub) {
      const bf16* pa = &As[buf][mi2 * 32 + sub * 16 + col16][hi ? 16 : 0];
      v16bf a = frag16(pa, pa + 8);
      acc[sub] = __builtin_amdgcn_wmma_f32_16x16x32_bf16(
          false, a, false, b, (short)0, acc[sub], false, false);
    }
  };

  // ---- pipelined main loop ----
  issue_A(0, 0);
  load_B(0);
  store_B(0);
#if USE_ASYNC_LDS
  asm volatile("s_wait_asynccnt 0x0" ::: "memory");
#endif
  __syncthreads();
  const int nk = I_DIM / BK;
  for (int k = 0; k < nk; ++k) {
    const int cur = k & 1;
    if (k + 1 < nk) {
      issue_A((k + 1) * BK, cur ^ 1);    // async fills next A buffer under wmma
      load_B((k + 1) * BK);              // weight loads in flight under wmma
    }
    compute(cur);
    if (k + 1 < nk) store_B(cur ^ 1);
#if USE_ASYNC_LDS
    asm volatile("s_wait_asynccnt 0x0" ::: "memory");
#endif
    __syncthreads();
  }

#pragma unroll
  for (int sub = 0; sub < 2; ++sub) {
#pragma unroll
    for (int r = 0; r < 8; ++r) {
      int m = mi2 * 32 + sub * 16 + r + hi * 8;
      if (m < rowsValid)
        out[(size_t)(rowStart + m) * H_DIM + colBase + ni * 16 + col16] = acc[sub][r];
    }
  }
}

// ------------------------------- launcher ----------------------------------
extern "C" void kernel_launch(void* const* d_in, const int* in_sizes, int n_in,
                              void* d_out, int out_size, void* d_ws, size_t ws_size,
                              hipStream_t stream) {
  const float* x      = (const float*)d_in[0];   // [T, H]
  const float* wgu    = (const float*)d_in[1];   // [E, H, 2I]
  const float* wd     = (const float*)d_in[2];   // [E, I, H]
  const int*   counts = (const int*)d_in[3];     // [E]
  bf16*  hidden = (bf16*)d_ws;                   // [T, I] bf16, 4 MB
  float* out    = (float*)d_out;                 // [T, H]

  const int rowTiles = T_TOK / BM + N_EXP;       // >= sum(ceil(cnt_e/BM))

  dim3 g1(I_DIM / BN, rowTiles);
  k_gate_up_silu<<<g1, 256, 0, stream>>>(x, wgu, counts, hidden);

  dim3 g2(H_DIM / BN, rowTiles);
  k_down<<<g2, 256, 0, stream>>>(hidden, wd, counts, out);
}
